// DynamicMemory_42425686950304
// MI455X (gfx1250) — matrix operand order repaired and checked by reading, not compiled, not run on hardware
//
#include <hip/hip_runtime.h>
#include <hip/hip_bf16.h>
#include <math.h>

#define N_BLOCKS 20
#define BATCH    8192
#define EMBED    256
#define EPS      1e-8f

typedef __attribute__((ext_vector_type(16))) _Float16 v16h;
typedef __attribute__((ext_vector_type(8)))  _Float16 v8h;
typedef __attribute__((ext_vector_type(8)))  float    v8f;
typedef __attribute__((ext_vector_type(4)))  float    v4f;

// ---------------------------------------------------------------------------
// Split-f16 fp32 emulation:  v = (float)hi + (float)lo, 3 WMMAs ~ fp32 GEMM.
// ---------------------------------------------------------------------------
__device__ __forceinline__ v8f wmma3(v16h a_hi, v16h a_lo, v16h b_hi, v16h b_lo, v8f c) {
    c = __builtin_amdgcn_wmma_f32_16x16x32_f16(false, a_hi, false, b_hi, (short)0, c, false, false);
    c = __builtin_amdgcn_wmma_f32_16x16x32_f16(false, a_hi, false, b_lo, (short)0, c, false, false);
    c = __builtin_amdgcn_wmma_f32_16x16x32_f16(false, a_lo, false, b_hi, (short)0, c, false, false);
    return c;
}

// A-matrix fragment (16x32, f16) from a row-major f32 [*, EMBED] source,
// converted on the fly into hi/lo split halves.
// Lane l: m = rowBase + (l & 15), half = l >> 4.
// K elements held: {8*half + 0..7} and {16 + 8*half + 0..7}  (ISA 16-bit A layout).
__device__ __forceinline__ void load_a_frag(const float* __restrict__ src, int row, int kbase,
                                            int lane, v16h& hi, v16h& lo) {
    const int half = lane >> 4;
    const float* p = src + (size_t)row * EMBED + kbase + half * 8;
    v4f a0 = *(const v4f*)(p);        // K = 8*half + 0..3
    v4f a1 = *(const v4f*)(p + 4);    // K = 8*half + 4..7
    v4f a2 = *(const v4f*)(p + 16);   // K = 16 + 8*half + 0..3
    v4f a3 = *(const v4f*)(p + 20);   // K = 16 + 8*half + 4..7
    float buf[16];
#pragma unroll
    for (int j = 0; j < 4; ++j) {
        buf[j]      = a0[j];
        buf[4 + j]  = a1[j];
        buf[8 + j]  = a2[j];
        buf[12 + j] = a3[j];
    }
#pragma unroll
    for (int j = 0; j < 16; ++j) {
        _Float16 hv = (_Float16)buf[j];
        hi[j] = hv;
        lo[j] = (_Float16)(buf[j] - (float)hv);
    }
}

// B-matrix fragment (32x16, f16) from pre-split f16 weight [f, e] row-major.
// Lane l: column n = fbase + (l & 15), half = l >> 4 holds K = {8*half + 0..7} and {16 + 8*half + 0..7}.
// B = W^T, so lane n reads W[f=n, e=kbase+8*half .. +7] and W[f=n, e=kbase+16+8*half .. +7]: two 16B chunks.
__device__ __forceinline__ v16h load_b_frag(const _Float16* __restrict__ W, int fbase, int kbase,
                                            int lane) {
    const _Float16* p = W + (size_t)(fbase + (lane & 15)) * EMBED + kbase + (lane >> 4) * 8;
    v8h lo8 = *(const v8h*)(p);
    v8h hi8 = *(const v8h*)(p + 16);
    v16h r;
#pragma unroll
    for (int j = 0; j < 8; ++j) { r[j] = lo8[j]; r[8 + j] = hi8[j]; }
    return r;
}

// ---------------------------------------------------------------------------
// Kernel 0: split Wu, Wv, Ww (f32 [E,E]) into f16 hi/lo pairs in workspace.
// Layout: slot 0/1 = Wu_hi/lo, 2/3 = Wv_hi/lo, 4/5 = Ww_hi/lo.
// ---------------------------------------------------------------------------
__global__ __launch_bounds__(256) void k_convert(const float* __restrict__ Wu,
                                                 const float* __restrict__ Wv,
                                                 const float* __restrict__ Ww,
                                                 _Float16* __restrict__ out) {
    int i = blockIdx.x * blockDim.x + threadIdx.x;
    if (i >= EMBED * EMBED) return;
    const float* srcs[3] = {Wu, Wv, Ww};
#pragma unroll
    for (int m = 0; m < 3; ++m) {
        float v = srcs[m][i];
        _Float16 hv = (_Float16)v;
        out[(size_t)(2 * m + 0) * (EMBED * EMBED) + i] = hv;
        out[(size_t)(2 * m + 1) * (EMBED * EMBED) + i] = (_Float16)(v - (float)hv);
    }
}

// ---------------------------------------------------------------------------
// Kernel 1: xw[n,f] = sum_e x[n,e] * Ww[f,e] + bias[f]   (computed once).
// 8 waves/WG: 4 row-groups x 2 col-groups; wave tile = 16 rows x 128 cols.
// ---------------------------------------------------------------------------
__global__ __launch_bounds__(256) void k_xw(const float* __restrict__ x,
                                            const _Float16* __restrict__ wsplit,
                                            const float* __restrict__ bias,
                                            float* __restrict__ xw) {
    const int lane = threadIdx.x & 31;
    const int wave = threadIdx.x >> 5;
    const int rowBase = blockIdx.x * 64 + (wave & 3) * 16;
    const int colBase = (wave >> 2) * 128;
    const _Float16* Whi = wsplit + (size_t)4 * EMBED * EMBED;
    const _Float16* Wlo = wsplit + (size_t)5 * EMBED * EMBED;

    v8f acc[8];
#pragma unroll
    for (int nt = 0; nt < 8; ++nt)
#pragma unroll
        for (int r = 0; r < 8; ++r) acc[nt][r] = 0.0f;

    const int row = rowBase + (lane & 15);
    for (int kt = 0; kt < 8; ++kt) {
        const int kbase = kt * 32;
        v16h ahi, alo;
        load_a_frag(x, row, kbase, lane, ahi, alo);
#pragma unroll
        for (int nt = 0; nt < 8; ++nt) {
            v16h bhi = load_b_frag(Whi, colBase + nt * 16, kbase, lane);
            v16h blo = load_b_frag(Wlo, colBase + nt * 16, kbase, lane);
            acc[nt] = wmma3(ahi, alo, bhi, blo, acc[nt]);
        }
    }

    const int halfm = lane >> 4;
#pragma unroll
    for (int nt = 0; nt < 8; ++nt) {
        const int f = colBase + nt * 16 + (lane & 15);
        const float bv = bias[f];
#pragma unroll
        for (int r = 0; r < 8; ++r) {
            const int m = rowBase + halfm * 8 + r;
            xw[(size_t)m * EMBED + f] = acc[nt][r] + bv;
        }
    }
}

// ---------------------------------------------------------------------------
// Kernel 2: per (block, 64-row tile):
//   pre  = h@Wu^T + w@Wv^T + xw                    (split-f16 WMMA, f32 accum)
//   gate = sigmoid(x*(h+w)); ht = PReLU(pre, a)
//   h_i  = h + gate*ht;  out = h_i / (rowsum(h_i) + EPS)
// ---------------------------------------------------------------------------
__global__ __launch_bounds__(256) void k_main(const float* __restrict__ x,
                                              const float* __restrict__ h,
                                              const float* __restrict__ w,
                                              const _Float16* __restrict__ wsplit,
                                              const float* __restrict__ xw,
                                              const float* __restrict__ a_slope,
                                              float* __restrict__ out) {
    __shared__ float s_rowsum[64];

    const int lane = threadIdx.x & 31;
    const int wave = threadIdx.x >> 5;
    const int rowGroup = wave & 3;
    const int colGroup = wave >> 2;
    const int rowBase = blockIdx.x * 64 + rowGroup * 16;
    const int colBase = colGroup * 128;
    const int blk = blockIdx.y;
    const float slope = a_slope[0];

    const _Float16* Wu_hi = wsplit;
    const _Float16* Wu_lo = wsplit + (size_t)1 * EMBED * EMBED;
    const _Float16* Wv_hi = wsplit + (size_t)2 * EMBED * EMBED;
    const _Float16* Wv_lo = wsplit + (size_t)3 * EMBED * EMBED;

    const float* hB = h + (size_t)blk * BATCH * EMBED;
    const float* wB = w + (size_t)blk * BATCH * EMBED;

    if (threadIdx.x < 64) s_rowsum[threadIdx.x] = EPS;  // denom = sum + EPS
    __syncthreads();

    v8f acc[8];
#pragma unroll
    for (int nt = 0; nt < 8; ++nt)
#pragma unroll
        for (int r = 0; r < 8; ++r) acc[nt][r] = 0.0f;

    const int row = rowBase + (lane & 15);
    for (int kt = 0; kt < 8; ++kt) {
        const int kbase = kt * 32;
        if (kt < 7) {  // hint the streamed A operands for the next K tile
            __builtin_prefetch(hB + (size_t)row * EMBED + kbase + 32, 0, 0);
            __builtin_prefetch(wB + (size_t)row * EMBED + kbase + 32, 0, 0);
        }
        v16h hhi, hlo, uhi, ulo;
        load_a_frag(hB, row, kbase, lane, hhi, hlo);
        load_a_frag(wB, row, kbase, lane, uhi, ulo);
#pragma unroll
        for (int nt = 0; nt < 8; ++nt) {
            const int fb = colBase + nt * 16;
            v16h bu_hi = load_b_frag(Wu_hi, fb, kbase, lane);
            v16h bu_lo = load_b_frag(Wu_lo, fb, kbase, lane);
            acc[nt] = wmma3(hhi, hlo, bu_hi, bu_lo, acc[nt]);
            v16h bv_hi = load_b_frag(Wv_hi, fb, kbase, lane);
            v16h bv_lo = load_b_frag(Wv_lo, fb, kbase, lane);
            acc[nt] = wmma3(uhi, ulo, bv_hi, bv_lo, acc[nt]);
        }
    }

    // Fused elementwise epilogue; acc[nt][r] -> h_i, accumulate row sums.
    const int halfm = lane >> 4;
    float rsum[8];
#pragma unroll
    for (int r = 0; r < 8; ++r) rsum[r] = 0.0f;

#pragma unroll
    for (int nt = 0; nt < 8; ++nt) {
        const int f = colBase + nt * 16 + (lane & 15);
#pragma unroll
        for (int r = 0; r < 8; ++r) {
            const int m = rowBase + halfm * 8 + r;
            const size_t idx = (size_t)m * EMBED + f;
            const float pre = acc[nt][r] + xw[idx];
            const float ht = (pre >= 0.0f) ? pre : slope * pre;
            const float xv = x[idx];
            const float hv = hB[idx];
            const float wv = wB[idx];
            const float gate = 1.0f / (1.0f + __expf(-(xv * (hv + wv))));
            const float hi_val = hv + gate * ht;
            acc[nt][r] = hi_val;
            rsum[r] += hi_val;
        }
    }

    // Reduce over the 16 lanes of each half (128 cols of this col-group).
#pragma unroll
    for (int r = 0; r < 8; ++r) {
        float v = rsum[r];
        v += __shfl_xor(v, 1, 32);
        v += __shfl_xor(v, 2, 32);
        v += __shfl_xor(v, 4, 32);
        v += __shfl_xor(v, 8, 32);
        rsum[r] = v;
    }
    if ((lane & 15) == 0) {  // lane 0 (rows r) and lane 16 (rows 8+r) contribute
#pragma unroll
        for (int r = 0; r < 8; ++r)
            atomicAdd(&s_rowsum[rowGroup * 16 + halfm * 8 + r], rsum[r]);
    }
    __syncthreads();

    float* outB = out + (size_t)blk * BATCH * EMBED;
#pragma unroll
    for (int r = 0; r < 8; ++r) {
        const int m = rowBase + halfm * 8 + r;
        const float inv = 1.0f / s_rowsum[rowGroup * 16 + halfm * 8 + r];
#pragma unroll
        for (int nt = 0; nt < 8; ++nt) {
            const int f = colBase + nt * 16 + (lane & 15);
            outB[(size_t)m * EMBED + f] = acc[nt][r] * inv;
        }
    }
}

// ---------------------------------------------------------------------------
extern "C" void kernel_launch(void* const* d_in, const int* in_sizes, int n_in,
                              void* d_out, int out_size, void* d_ws, size_t ws_size,
                              hipStream_t stream) {
    const float* x    = (const float*)d_in[0];
    const float* h    = (const float*)d_in[1];
    const float* w    = (const float*)d_in[2];
    const float* Wu   = (const float*)d_in[3];
    const float* Wv   = (const float*)d_in[4];
    const float* Ww   = (const float*)d_in[5];
    const float* bias = (const float*)d_in[6];
    const float* a    = (const float*)d_in[7];
    float* out = (float*)d_out;

    // Workspace: 6 x [E*E] f16 split weights (768 KB), then xw f32 [BATCH*E] (8 MB).
    _Float16* wsplit = (_Float16*)d_ws;
    float* xw = (float*)((char*)d_ws + (size_t)6 * EMBED * EMBED * sizeof(_Float16));

    k_convert<<<(EMBED * EMBED + 255) / 256, 256, 0, stream>>>(Wu, Wv, Ww, wsplit);
    k_xw<<<BATCH / 64, 256, 0, stream>>>(x, wsplit, bias, xw);
    dim3 grid(BATCH / 64, N_BLOCKS);
    k_main<<<grid, 256, 0, stream>>>(x, h, w, wsplit, xw, a, out);

    (void)in_sizes; (void)n_in; (void)out_size; (void)ws_size;
}
